// WindowAttention_64149631533662
// MI455X (gfx1250) — compile-verified, matrix operand order checked by reading, not run on hardware
//
#include <hip/hip_runtime.h>
#include <hip/hip_bf16.h>
#include <stddef.h>

// ---------------------------------------------------------------------------
// Swin window attention, fully fused for MI455X (gfx1250, wave32, WMMA bf16).
//   DIM=384, HEADS=12, HEAD_DIM=32, N=49 (padded to 64), B=4096, NW=64
// One kernel: x --async-to-LDS--> qkv GEMM -> attention+softmax -> P@V ->
// output projection, all in LDS. HBM traffic = read x + write out only.
// ---------------------------------------------------------------------------

#define DIMC   384
#define DIM3   1152
#define HEADS  12
#define HD     32
#define NTOK   49
#define NPAD   64
#define NWIN   64
#define BWIN   4096

// LDS byte offsets (total 294912 B = 288 KB <= 320 KB WGP LDS)
#define SM_R0     0        // raw x fp32 (75264 B) -> P bf16 (98304 B) -> proj A-frags (49152 B)
#define SM_QKV    98304    // qkv row-major bf16 [64][1152] = 147456 B
#define SM_R2     245760   // X A-frags (49152 B) -> attended row-major bf16 [64][384] (49152 B)
#define SM_TOTAL  294912

typedef __attribute__((ext_vector_type(16))) __bf16 v16bf;
typedef __attribute__((ext_vector_type(8)))  float  v8f;

union Frag16 {
    unsigned short u[16];
    v16bf          v;
};

// fp32 -> bf16 round-to-nearest-even (bit trick)
__device__ __forceinline__ unsigned short f2bf(float f) {
    unsigned int u = __float_as_uint(f);
    unsigned int r = u + 0x7FFFu + ((u >> 16) & 1u);
    return (unsigned short)(r >> 16);
}

// 16-bit A/B fragment K mapping (ISA 7.12.2, 16-bit A-matrix 16x32):
//   lanes 0-15 hold K = {0..7, 16..23}, lanes 16-31 hold K = {8..15, 24..31}
__device__ __forceinline__ int kmap(int e, int lane) {
    return ((e < 8) ? e : (e + 8)) + ((lane >= 16) ? 8 : 0);
}

__device__ __forceinline__ float rmax16(float v) {
#pragma unroll
    for (int m = 1; m < 16; m <<= 1) v = fmaxf(v, __shfl_xor(v, m, 32));
    return v;
}
__device__ __forceinline__ float rsum16(float v) {
#pragma unroll
    for (int m = 1; m < 16; m <<= 1) v += __shfl_xor(v, m, 32);
    return v;
}

// ---------------------------------------------------------------------------
// Prep kernels: weight swizzle to WMMA B-fragment layout, bias, combined mask
// ---------------------------------------------------------------------------

// wfrag[kc][nt][lane][e], kc: K-chunk of 32, nt: 16-col tile. q-scale folded in.
__global__ void prep_qkv_wfrag(const float* __restrict__ qkv_w,
                               unsigned short* __restrict__ wfrag) {
    int idx = blockIdx.x * blockDim.x + threadIdx.x;
    if (idx >= 12 * 72 * 32 * 16) return;
    int e    = idx & 15;
    int lane = (idx >> 4) & 31;
    int nt   = (idx >> 9) % 72;
    int kc   = idx / (512 * 72);
    int K    = kc * 32 + kmap(e, lane);
    int col  = nt * 16 + (lane & 15);
    float w = qkv_w[(size_t)K * DIM3 + col];
    if (col < DIMC) w *= 0.17677669529663687f;  // HEAD_DIM^-0.5 folded into q
    wfrag[idx] = f2bf(w);
}

__global__ void prep_proj_wfrag(const float* __restrict__ proj_w,
                                unsigned short* __restrict__ wfrag) {
    int idx = blockIdx.x * blockDim.x + threadIdx.x;
    if (idx >= 12 * 24 * 32 * 16) return;
    int e    = idx & 15;
    int lane = (idx >> 4) & 31;
    int nt   = (idx >> 9) % 24;
    int kc   = idx / (512 * 24);
    int K    = kc * 32 + kmap(e, lane);
    int col  = nt * 16 + (lane & 15);
    wfrag[idx] = f2bf(proj_w[(size_t)K * DIMC + col]);
}

__global__ void prep_qkvb(const float* __restrict__ qkv_b,
                          float* __restrict__ qkvb_s) {
    int c = blockIdx.x * blockDim.x + threadIdx.x;
    if (c >= DIM3) return;
    float v = qkv_b[c];
    if (c < DIMC) v *= 0.17677669529663687f;
    qkvb_s[c] = v;
}

// comb[w][h][i][j] (padded 64x64) = bias_table[rel_index[i,j]][h] + mask[w,i,j]
// pad cells = -1e30 so softmax zeroes them.
__global__ void prep_comb(const float* __restrict__ mask,
                          const int* __restrict__ rel_index,
                          const float* __restrict__ bias_table,
                          float* __restrict__ comb) {
    int idx = blockIdx.x * blockDim.x + threadIdx.x;
    if (idx >= NWIN * HEADS * NPAD * NPAD) return;
    int j = idx & 63;
    int i = (idx >> 6) & 63;
    int h = (idx >> 12) % HEADS;
    int w = idx / (4096 * HEADS);
    float v = -1e30f;
    if (i < NTOK && j < NTOK) {
        int r = rel_index[i * NTOK + j];
        v = bias_table[r * HEADS + h] + mask[((size_t)w * NTOK + i) * NTOK + j];
    }
    comb[idx] = v;
}

// ---------------------------------------------------------------------------
// Fully fused window attention block. One block per window; 12 waves.
// ---------------------------------------------------------------------------
__global__ __launch_bounds__(384, 1) void fused_window_attn(
    const float* __restrict__ x,               // [B][49][384]
    const unsigned short* __restrict__ wfrag,  // [12][72][32][16] bf16 (qkv_w)
    const float* __restrict__ qkvb,            // [1152] (q-scaled)
    const float* __restrict__ comb,            // [64][12][64][64]
    const unsigned short* __restrict__ pwfrag, // [12][24][32][16] bf16 (proj_w)
    const float* __restrict__ proj_b,          // [384]
    float* __restrict__ out)                   // [B][49][384] f32
{
    extern __shared__ char smem[];
    float*          xraw    = (float*)(smem + SM_R0);           // phase 1
    unsigned short* Plds    = (unsigned short*)(smem + SM_R0);  // phase 3
    unsigned short* Afrag   = (unsigned short*)(smem + SM_R0);  // phase 4
    unsigned short* qkv     = (unsigned short*)(smem + SM_QKV); // phases 2-3
    unsigned short* Xfrag   = (unsigned short*)(smem + SM_R2);  // phases 1-2
    unsigned short* att_lds = (unsigned short*)(smem + SM_R2);  // phases 3-4

    const int b    = blockIdx.x;
    const int tid  = threadIdx.x;
    const int wave = tid >> 5;
    const int lane = tid & 31;

    // ---- phase 1a: async-copy this window's x slab (75264 B, contiguous)
    //                straight into LDS via the CDNA5 async-to-LDS path -------
    {
        const float* xb = x + (size_t)b * NTOK * DIMC;
        const int nchunks = (NTOK * DIMC * 4) / 16;  // 4704 x b128
        for (int idx = tid; idx < nchunks; idx += 384) {
            unsigned int loff = (unsigned int)idx * 16u;  // LDS byte addr (R0)
            unsigned int goff = (unsigned int)idx * 16u;  // byte offset from xb
            asm volatile("global_load_async_to_lds_b128 %0, %1, %2"
                         :: "v"(loff), "v"(goff), "s"(xb)
                         : "memory");
        }
        asm volatile("s_wait_asynccnt 0x0" ::: "memory");
    }
    __syncthreads();

    // ---- phase 1b: build X A-fragments (fp32 -> bf16) from LDS ------------
    for (int idx = tid; idx < 4 * 12 * 32 * 16; idx += 384) {
        int e  = idx & 15;
        int ln = (idx >> 4) & 31;
        int kc = (idx >> 9) % 12;
        int mt = idx / (512 * 12);
        int row = mt * 16 + (ln & 15);
        int k   = kc * 32 + kmap(e, ln);
        float v = (row < NTOK) ? xraw[row * DIMC + k] : 0.0f;
        Xfrag[idx] = f2bf(v);
    }
    __syncthreads();

    // ---- phase 2: qkv = X @ W + b  (288 16x16 tiles over 12 waves) --------
    for (int t = wave; t < 288; t += 12) {
        int mt  = t / 72;
        int nt  = t % 72;
        int col = nt * 16 + (lane & 15);
        float cb = qkvb[col];
        v8f C = {cb, cb, cb, cb, cb, cb, cb, cb};
#pragma unroll
        for (int kc = 0; kc < 12; ++kc) {
            if (kc + 1 < 12)
                __builtin_prefetch(&wfrag[(((size_t)(kc + 1) * 72 + nt) * 32 + lane) * 16], 0, 3);
            v16bf a  = *(const v16bf*)&Xfrag[((mt * 12 + kc) * 32 + lane) * 16];
            v16bf bm = *(const v16bf*)&wfrag[(((size_t)kc * 72 + nt) * 32 + lane) * 16];
            C = __builtin_amdgcn_wmma_f32_16x16x32_bf16(false, a, false, bm,
                                                        (short)0, C, false, false);
        }
        int rbase = mt * 16 + 8 * (lane >> 4);  // C layout: vgpr r -> row r (+8 hi half)
#pragma unroll
        for (int r = 0; r < 8; ++r)
            qkv[(rbase + r) * DIM3 + col] = f2bf(C[r]);
    }
    __syncthreads();

    // ---- phase 3: attention, one head per wave ----------------------------
    const int h = wave;

    // gather q A-fragments (K = head_dim = 32 -> single WMMA K step)
    Frag16 qa[4];
#pragma unroll
    for (int mt = 0; mt < 4; ++mt)
#pragma unroll
        for (int e = 0; e < 16; ++e) {
            int row = mt * 16 + (lane & 15);
            int d   = kmap(e, lane);
            qa[mt].u[e] = qkv[row * DIM3 + h * HD + d];
        }

    v8f A[4][4];
#pragma unroll
    for (int nt = 0; nt < 4; ++nt) {
        Frag16 kb;
#pragma unroll
        for (int e = 0; e < 16; ++e) {
            int j = nt * 16 + (lane & 15);
            int d = kmap(e, lane);
            kb.u[e] = qkv[j * DIM3 + DIMC + h * HD + d];
        }
#pragma unroll
        for (int mt = 0; mt < 4; ++mt) {
            v8f z = {};
            A[mt][nt] = __builtin_amdgcn_wmma_f32_16x16x32_bf16(
                false, qa[mt].v, false, kb.v, (short)0, z, false, false);
        }
    }

    // add relative-position bias + window mask (precombined, padded with -1e30)
    const float* cbp = comb + (((size_t)(b % NWIN) * HEADS + h) * NPAD) * NPAD;
#pragma unroll
    for (int mt = 0; mt < 4; ++mt)
#pragma unroll
        for (int nt = 0; nt < 4; ++nt)
#pragma unroll
            for (int r = 0; r < 8; ++r) {
                int i = mt * 16 + r + 8 * (lane >> 4);
                int j = nt * 16 + (lane & 15);
                A[mt][nt][r] += cbp[i * NPAD + j];
            }

    // softmax over j: each row lives in one 16-lane half at fixed vgpr index
#pragma unroll
    for (int mt = 0; mt < 4; ++mt)
#pragma unroll
        for (int r = 0; r < 8; ++r) {
            float m = A[mt][0][r];
#pragma unroll
            for (int nt = 1; nt < 4; ++nt) m = fmaxf(m, A[mt][nt][r]);
            m = rmax16(m);
            float s = 0.0f;
#pragma unroll
            for (int nt = 0; nt < 4; ++nt) {
                float ev = __expf(A[mt][nt][r] - m);
                A[mt][nt][r] = ev;
                s += ev;
            }
            s = rsum16(s);
            float inv = 1.0f / s;
#pragma unroll
            for (int nt = 0; nt < 4; ++nt) A[mt][nt][r] *= inv;
        }

    // write P (bf16, row-major) to this head's LDS region
    unsigned short* P = Plds + h * NPAD * NPAD;
#pragma unroll
    for (int mt = 0; mt < 4; ++mt)
#pragma unroll
        for (int nt = 0; nt < 4; ++nt)
#pragma unroll
            for (int r = 0; r < 8; ++r) {
                int i = mt * 16 + r + 8 * (lane >> 4);
                int j = nt * 16 + (lane & 15);
                P[i * NPAD + j] = f2bf(A[mt][nt][r]);
            }
    __syncthreads();

    // out_head = P (64x64) @ V (64x32), written row-major bf16 to att_lds
    Frag16 pa[4][2];
#pragma unroll
    for (int mt = 0; mt < 4; ++mt)
#pragma unroll
        for (int jc = 0; jc < 2; ++jc)
#pragma unroll
            for (int e = 0; e < 16; ++e) {
                int i = mt * 16 + (lane & 15);
                int j = jc * 32 + kmap(e, lane);
                pa[mt][jc].u[e] = P[i * NPAD + j];
            }
#pragma unroll
    for (int dt = 0; dt < 2; ++dt) {
        Frag16 vb[2];
#pragma unroll
        for (int jc = 0; jc < 2; ++jc)
#pragma unroll
            for (int e = 0; e < 16; ++e) {
                int d = dt * 16 + (lane & 15);
                int j = jc * 32 + kmap(e, lane);
                vb[jc].u[e] = qkv[j * DIM3 + 2 * DIMC + h * HD + d];
            }
#pragma unroll
        for (int mt = 0; mt < 4; ++mt) {
            v8f O = {};
#pragma unroll
            for (int jc = 0; jc < 2; ++jc)
                O = __builtin_amdgcn_wmma_f32_16x16x32_bf16(
                    false, pa[mt][jc].v, false, vb[jc].v, (short)0, O, false, false);
            int rbase = mt * 16 + 8 * (lane >> 4);
            int d     = dt * 16 + (lane & 15);
#pragma unroll
            for (int r = 0; r < 8; ++r)
                att_lds[(rbase + r) * DIMC + h * HD + d] = f2bf(O[r]);
        }
    }
    __syncthreads();

    // ---- phase 4a: rebuild attended values as proj A-fragments (R0) -------
    for (int idx = tid; idx < 4 * 12 * 32 * 16; idx += 384) {
        int e  = idx & 15;
        int ln = (idx >> 4) & 31;
        int kc = (idx >> 9) % 12;
        int mt = idx / (512 * 12);
        int row = mt * 16 + (ln & 15);
        int k   = kc * 32 + kmap(e, ln);
        Afrag[idx] = (row < NTOK) ? att_lds[row * DIMC + k] : (unsigned short)0;
    }
    __syncthreads();

    // ---- phase 4b: out = attended @ proj_w + proj_b (96 tiles, 12 waves) --
    for (int t = wave; t < 96; t += 12) {
        int mt  = t / 24;
        int nt  = t % 24;
        int col = nt * 16 + (lane & 15);
        float cb = proj_b[col];
        v8f C = {cb, cb, cb, cb, cb, cb, cb, cb};
#pragma unroll
        for (int kc = 0; kc < 12; ++kc) {
            if (kc + 1 < 12)
                __builtin_prefetch(&pwfrag[(((size_t)(kc + 1) * 24 + nt) * 32 + lane) * 16], 0, 3);
            v16bf a  = *(const v16bf*)&Afrag[((mt * 12 + kc) * 32 + lane) * 16];
            v16bf bm = *(const v16bf*)&pwfrag[(((size_t)kc * 24 + nt) * 32 + lane) * 16];
            C = __builtin_amdgcn_wmma_f32_16x16x32_bf16(false, a, false, bm,
                                                        (short)0, C, false, false);
        }
        int rbase = mt * 16 + 8 * (lane >> 4);
#pragma unroll
        for (int r = 0; r < 8; ++r) {
            int i = rbase + r;
            if (i < NTOK)
                out[(size_t)b * NTOK * DIMC + (size_t)i * DIMC + col] = C[r];
        }
    }
}

// ---------------------------------------------------------------------------
// Launcher
// ---------------------------------------------------------------------------
extern "C" void kernel_launch(void* const* d_in, const int* in_sizes, int n_in,
                              void* d_out, int out_size, void* d_ws, size_t ws_size,
                              hipStream_t stream) {
    const float* x      = (const float*)d_in[0];
    const float* mask   = (const float*)d_in[1];
    const int*   rel    = (const int*)d_in[2];
    const float* qkv_w  = (const float*)d_in[3];
    const float* qkv_b  = (const float*)d_in[4];
    const float* proj_w = (const float*)d_in[5];
    const float* proj_b = (const float*)d_in[6];
    const float* btab   = (const float*)d_in[7];
    float*       out    = (float*)d_out;

    char* ws = (char*)d_ws;
    // workspace layout (bytes), total ~13.8 MB:
    unsigned short* wqkv   = (unsigned short*)(ws + 0);       //   884736
    unsigned short* wproj  = (unsigned short*)(ws + 884736);  //   294912
    float*          qkvb_s = (float*)(ws + 1179648);          //     4608
    float*          comb   = (float*)(ws + 1184256);          // 12582912

    prep_qkv_wfrag<<<(12 * 72 * 512 + 255) / 256, 256, 0, stream>>>(qkv_w, wqkv);
    prep_proj_wfrag<<<(12 * 24 * 512 + 255) / 256, 256, 0, stream>>>(proj_w, wproj);
    prep_qkvb<<<(DIM3 + 255) / 256, 256, 0, stream>>>(qkv_b, qkvb_s);
    prep_comb<<<(NWIN * HEADS * NPAD * NPAD + 255) / 256, 256, 0, stream>>>(
        mask, rel, btab, comb);

    fused_window_attn<<<BWIN, 384, SM_TOTAL, stream>>>(
        x, wqkv, qkvb_s, comb, wproj, proj_b, out);
}
